// SFSRec_14448269983910
// MI455X (gfx1250) — matrix operand order repaired and checked by reading, not compiled, not run on hardware
//
#include <hip/hip_runtime.h>
#include <math.h>

typedef unsigned short u16;
typedef __attribute__((ext_vector_type(16))) __bf16 v16bf;
typedef __attribute__((ext_vector_type(8)))  float  v8f;

#define LN_EPS 1e-8f

__device__ __forceinline__ u16 f2bf(float f) {
    unsigned int u = __float_as_uint(f);
    unsigned int r = u + 0x7FFFu + ((u >> 16) & 1u);   // round-to-nearest-even
    return (u16)(r >> 16);
}

// ---------------------------------------------------------------- pack fp32 -> bf16
__global__ void sfs_pack_bf16(const float* __restrict__ s, u16* __restrict__ d, int n) {
    for (int i = blockIdx.x * blockDim.x + threadIdx.x; i < n; i += gridDim.x * blockDim.x)
        d[i] = f2bf(s[i]);
}
__global__ void sfs_pack_bf16_pad(const float* __restrict__ s, u16* __restrict__ d,
                                  int nsrc, int ntot) {
    for (int i = blockIdx.x * blockDim.x + threadIdx.x; i < ntot; i += gridDim.x * blockDim.x)
        d[i] = (i < nsrc) ? f2bf(s[i]) : (u16)0;
}

// ---------------------------------------------------------------- embedding gather
// pre[row,h] = item_emb[ids[row], h] * sqrt(H) + pos_emb[row % L, h]
__global__ void sfs_embed(const int* __restrict__ ids,
                          const float* __restrict__ item_emb,
                          const float* __restrict__ pos_emb,
                          float* __restrict__ pre, int L, int H, float scale) {
    const int row = blockIdx.x;
    const int h   = threadIdx.x;
    const int id  = ids[row];
    const int l   = row % L;
    pre[(size_t)row * H + h] = item_emb[(size_t)id * H + h] * scale + pos_emb[(size_t)l * H + h];
}

// ---------------------------------------------------------------- fused LN(+res)+mask
// t = in + (res?res:0); y = LN(t)*g+b; y *= (ids[row]!=0); write f32 + bf16
__global__ void sfs_ln(const float* __restrict__ in, const float* __restrict__ res,
                       const float* __restrict__ g, const float* __restrict__ b,
                       const int* __restrict__ ids,
                       float* __restrict__ outf, u16* __restrict__ outb, int H) {
    __shared__ float red[32];
    const int row  = blockIdx.x;
    const int h    = threadIdx.x;
    const int lane = h & 31;
    const int wid  = h >> 5;
    const int nw   = blockDim.x >> 5;
    const size_t idx = (size_t)row * H + h;

    float t = in[idx] + (res ? res[idx] : 0.f);

    float s = t;
    #pragma unroll
    for (int o = 16; o > 0; o >>= 1) s += __shfl_down(s, o, 32);
    if (lane == 0) red[wid] = s;
    __syncthreads();
    float tot = 0.f;
    for (int i = 0; i < nw; i++) tot += red[i];
    const float mean = tot / (float)H;
    __syncthreads();

    const float d = t - mean;
    float s2 = d * d;
    #pragma unroll
    for (int o = 16; o > 0; o >>= 1) s2 += __shfl_down(s2, o, 32);
    if (lane == 0) red[wid] = s2;
    __syncthreads();
    float tot2 = 0.f;
    for (int i = 0; i < nw; i++) tot2 += red[i];
    const float var = tot2 / (float)H;

    float y = d * rsqrtf(var + LN_EPS) * g[h] + b[h];
    if (ids[row] == 0) y = 0.f;
    outf[idx] = y;
    outb[idx] = f2bf(y);
}

// ---------------------------------------------------------------- causal uniform avg
// m[b,l,h] = (sum_{j<=l} v[b,j,h]) / (l+1)
__global__ void sfs_causal_avg(const float* __restrict__ v, float* __restrict__ m,
                               int L, int H) {
    const int b = blockIdx.x;
    const int h = threadIdx.x;
    const float* vp = v + (size_t)b * L * H + h;
    float*       mp = m + (size_t)b * L * H + h;
    float s = 0.f;
    for (int l = 0; l < L; l++) {
        s += vp[(size_t)l * H];
        mp[(size_t)l * H] = s / (float)(l + 1);
    }
}

// ---------------------------------------------------------------- WMMA bf16 GEMM
// C[M,N] = A[M,K](bf16) * W[N,K]^T(bf16)  (+bias[n]) (ReLU)
// 128 threads = 4 waves; each wave computes a 32x64 tile (2 M-frags x 4 N-frags,
// 8 WMMAs per 32-wide K step -> 21.3 FLOP per cache byte); block tile 64x128.
// M must be a multiple of 64; W rows padded to a multiple of 128 so N-tile
// loads never go OOB; stores guarded by Nvalid.
__global__ __launch_bounds__(128)
void sfs_gemm_bf16(const u16* __restrict__ A, const u16* __restrict__ Bw,
                   const float* __restrict__ bias, int relu,
                   float* __restrict__ Cf, u16* __restrict__ Cb,
                   int K, int ldc, int Nvalid) {
    const int wave = threadIdx.x >> 5;
    const int lane = threadIdx.x & 31;
    const int half = lane >> 4;      // 0: lanes 0-15, 1: lanes 16-31
    const int lidx = lane & 15;
    const int mBase = blockIdx.x * 64  + (wave >> 1) * 32;
    const int nBase = blockIdx.y * 128 + (wave & 1) * 64;

    v8f acc[2][4] = {};

    const u16* arow0 = A + (size_t)(mBase + lidx) * K;
    const u16* arow1 = A + (size_t)(mBase + 16 + lidx) * K;
    const u16* brow[4];
    #pragma unroll
    for (int j = 0; j < 4; j++)
        brow[j] = Bw + (size_t)(nBase + j * 16 + lidx) * K;

    for (int k = 0; k < K; k += 32) {
        union { uint4 q[2]; v16bf v; } a0, a1, bf[4];
        // A 16x32 bf16 layout: lane<16 holds K[k..k+7],[k+16..k+23]; lane>=16 +8
        const int ka = k + half * 8;
        a0.q[0] = *(const uint4*)(arow0 + ka);
        a0.q[1] = *(const uint4*)(arow0 + ka + 16);
        a1.q[0] = *(const uint4*)(arow1 + ka);
        a1.q[1] = *(const uint4*)(arow1 + ka + 16);
        // B 32x16 bf16 layout: lane holds column (nTile+lidx), K[k+half*8 .. +7],[k+16+half*8 .. +7]
        const int kb = k + half * 8;
        #pragma unroll
        for (int j = 0; j < 4; j++) {
            bf[j].q[0] = *(const uint4*)(brow[j] + kb);
            bf[j].q[1] = *(const uint4*)(brow[j] + kb + 16);
        }
        #pragma unroll
        for (int j = 0; j < 4; j++) {
            acc[0][j] = __builtin_amdgcn_wmma_f32_16x16x32_bf16(false, a0.v, false, bf[j].v,
                                                                (short)0, acc[0][j], false, false);
            acc[1][j] = __builtin_amdgcn_wmma_f32_16x16x32_bf16(false, a1.v, false, bf[j].v,
                                                                (short)0, acc[1][j], false, false);
        }
    }

    // C/D layout: element r of acc -> M = mTile + half*8 + r, N = nTile + lidx
    #pragma unroll
    for (int i = 0; i < 2; i++) {
        #pragma unroll
        for (int j = 0; j < 4; j++) {
            const int gn = nBase + j * 16 + lidx;
            if (gn >= Nvalid) continue;
            const float bb = bias ? bias[gn] : 0.f;
            #pragma unroll
            for (int r = 0; r < 8; r++) {
                const int gm = mBase + i * 16 + half * 8 + r;
                float val = acc[i][j][r] + bb;
                if (relu && val < 0.f) val = 0.f;
                const size_t off = (size_t)gm * ldc + gn;
                if (Cf) Cf[off] = val;
                if (Cb) Cb[off] = f2bf(val);
            }
        }
    }
}

// ================================================================ launch
extern "C" void kernel_launch(void* const* d_in, const int* in_sizes, int n_in,
                              void* d_out, int out_size, void* d_ws, size_t ws_size,
                              hipStream_t stream) {
    const int*   ids      = (const int*)  d_in[0];
    const float* item_emb = (const float*)d_in[1];
    const float* pos_emb  = (const float*)d_in[2];
    const float* emb_g    = (const float*)d_in[3];
    const float* emb_b    = (const float*)d_in[4];
    const float* Wv       = (const float*)d_in[5];
    const float* Wo       = (const float*)d_in[6];
    const float* attn_g   = (const float*)d_in[7];
    const float* attn_b   = (const float*)d_in[8];
    const float* W1       = (const float*)d_in[9];
    const float* b1       = (const float*)d_in[10];
    const float* W2       = (const float*)d_in[11];
    const float* b2       = (const float*)d_in[12];
    const float* ffn_g    = (const float*)d_in[13];
    const float* ffn_b    = (const float*)d_in[14];

    const int H  = in_sizes[3];             // 256
    const int R  = in_sizes[0];             // B*L = 3200
    const int L  = in_sizes[2] / H;         // 200
    const int Bn = R / L;                   // 16
    const int V1 = in_sizes[1] / H;         // V+1 = 50001
    const int NB = in_sizes[5] / (H * H);   // 2
    const int Vp = ((V1 + 127) / 128) * 128; // padded logits N (50048)

    // ---- workspace bump allocator (256 B aligned) ----
    char* p = (char*)d_ws;
    auto alloc = [&](size_t bytes) -> char* {
        char* r = p; p += (bytes + 255) & ~(size_t)255; return r;
    };
    float* xf  = (float*)alloc((size_t)R * H * sizeof(float));   // current x (f32)
    float* vf  = (float*)alloc((size_t)R * H * sizeof(float));   // pre-LN / proj out / ffn out
    float* mf  = (float*)alloc((size_t)R * H * sizeof(float));   // causal-avg result
    u16*   xb  = (u16*)  alloc((size_t)R * H * sizeof(u16));     // bf16 of x
    u16*   t1b = (u16*)  alloc((size_t)R * H * sizeof(u16));     // bf16 intermediate
    u16*   Wvb = (u16*)  alloc((size_t)NB * H * H * sizeof(u16));
    u16*   Wob = (u16*)  alloc((size_t)NB * H * H * sizeof(u16));
    u16*   W1b = (u16*)  alloc((size_t)NB * H * H * sizeof(u16));
    u16*   W2b = (u16*)  alloc((size_t)NB * H * H * sizeof(u16));
    u16*   Eb  = (u16*)  alloc((size_t)Vp * H * sizeof(u16));    // bf16 item_emb, zero-padded

    // ---- pack weights / embedding table to bf16 ----
    const int nw = NB * H * H;
    sfs_pack_bf16<<<256, 256, 0, stream>>>(Wv, Wvb, nw);
    sfs_pack_bf16<<<256, 256, 0, stream>>>(Wo, Wob, nw);
    sfs_pack_bf16<<<256, 256, 0, stream>>>(W1, W1b, nw);
    sfs_pack_bf16<<<256, 256, 0, stream>>>(W2, W2b, nw);
    sfs_pack_bf16_pad<<<4096, 256, 0, stream>>>(item_emb, Eb, V1 * H, Vp * H);

    // ---- embedding + first LN ----
    sfs_embed<<<R, H, 0, stream>>>(ids, item_emb, pos_emb, vf, L, H, sqrtf((float)H));
    sfs_ln<<<R, H, 0, stream>>>(vf, nullptr, emb_g, emb_b, ids, xf, xb, H);

    // ---- transformer blocks ----
    dim3 gh(R / 64, H / 128);           // 50 x 2 for hidden GEMMs
    for (int i = 0; i < NB; i++) {
        const size_t wo = (size_t)i * H * H;
        // v1 = x @ Wv^T   (bf16 out only)
        sfs_gemm_bf16<<<gh, 128, 0, stream>>>(xb, Wvb + wo, nullptr, 0,
                                              nullptr, t1b, H, H, H);
        // v  = v1 @ Wo^T  (f32 out)
        sfs_gemm_bf16<<<gh, 128, 0, stream>>>(t1b, Wob + wo, nullptr, 0,
                                              vf, nullptr, H, H, H);
        // mixed = cumsum(v)/counts
        sfs_causal_avg<<<Bn, H, 0, stream>>>(vf, mf, L, H);
        // x = LN(x + mixed) * keep
        sfs_ln<<<R, H, 0, stream>>>(xf, mf, attn_g + (size_t)i * H, attn_b + (size_t)i * H,
                                    ids, xf, xb, H);
        // y1 = relu(x @ W1^T + b1)   (bf16 out)
        sfs_gemm_bf16<<<gh, 128, 0, stream>>>(xb, W1b + wo, b1 + (size_t)i * H, 1,
                                              nullptr, t1b, H, H, H);
        // y  = y1 @ W2^T + b2        (f32 out)
        sfs_gemm_bf16<<<gh, 128, 0, stream>>>(t1b, W2b + wo, b2 + (size_t)i * H, 0,
                                              vf, nullptr, H, H, H);
        // x = LN(x + y) * keep
        sfs_ln<<<R, H, 0, stream>>>(xf, vf, ffn_g + (size_t)i * H, ffn_b + (size_t)i * H,
                                    ids, xf, xb, H);
    }

    // ---- logits = x @ item_emb^T  -> d_out (f32, ldc = V1, guarded columns) ----
    dim3 gl(R / 64, Vp / 128);          // 50 x 391
    sfs_gemm_bf16<<<gl, 128, 0, stream>>>(xb, Eb, nullptr, 0,
                                          (float*)d_out, nullptr, H, V1, V1);
}
